// EmbeddingLayer_78932908965942
// MI455X (gfx1250) — matrix operand order repaired and checked by reading, not compiled, not run on hardware
//
#include <hip/hip_runtime.h>

// ---------------------------------------------------------------------------
// out[b,d] = sum_f (indices[b,f] != 0) * W[f,d]
// BATCH=4096, FIELD=20000, EMBED=64.
// Bandwidth-bound: 328 MB of int32 indices @ 23.3 TB/s => ~14 us floor.
// Strategy: stream indices once, W repacked to f16 in WMMA-B layout (L2
// resident), v_wmma_f32_16x16x32_f16 with f32 accumulation, atomic combine.
// ---------------------------------------------------------------------------

typedef __attribute__((ext_vector_type(16))) _Float16 v16h;
typedef __attribute__((ext_vector_type(8)))  float    v8f;
typedef __attribute__((ext_vector_type(4)))  int      v4i;

#define BATCH   4096
#define FIELD   20000
#define EMBED   64
#define KCHUNK  32
#define NCHUNKS (FIELD / KCHUNK)          /* 625 */
#define NSPLIT  25
#define CPS     (NCHUNKS / NSPLIT)        /* 25 chunks (800 fields) per block */
#define W16_ELEMS ((size_t)NCHUNKS * 4 * 32 * 16)   /* 1,280,000 halfs = 2.56 MB */

// ---------------------------------------------------------------------------
// Repack W (f32 row-major [F,64]) -> f16 in WMMA B-operand order.
// B fragment for chunk c, N-tile n (columns 16n..16n+15), lane l holds 16
// halfs e=0..15 mapping to K = 32c + (l<16 ? 0 : 16) + e, N = 16n + (l&15).
// Stored contiguously: W16[(((c*4 + n)*32 + l)*16) + e].
// ---------------------------------------------------------------------------
__global__ void repack_w_kernel(const float* __restrict__ W,
                                _Float16* __restrict__ W16) {
    size_t tid = (size_t)blockIdx.x * blockDim.x + threadIdx.x;
    if (tid >= W16_ELEMS) return;
    int e = (int)(tid & 15);
    int l = (int)((tid >> 4) & 31);
    int n = (int)((tid >> 9) & 3);
    int c = (int)(tid >> 11);
    int k   = ((l < 16) ? 0 : 16) + e;
    int f   = c * KCHUNK + k;
    int col = n * 16 + (l & 15);
    W16[tid] = (_Float16)W[(size_t)f * EMBED + col];
}

__global__ void zero_out_kernel(float* __restrict__ out, int n) {
    int t = blockIdx.x * blockDim.x + threadIdx.x;
    if (t < n) out[t] = 0.0f;
}

// ---------------------------------------------------------------------------
// Main kernel: one wave32 per block. blockIdx.x selects a 16-row batch tile,
// blockIdx.y selects a K-split of CPS chunks. 4 accumulators cover N=64.
// ---------------------------------------------------------------------------
__global__ void __launch_bounds__(32)
embed_wmma_kernel(const int* __restrict__ idx,
                  const _Float16* __restrict__ W16,
                  float* __restrict__ out) {
    const int lane    = threadIdx.x;        // 0..31, full wave, EXEC all-ones
    const int m       = lane & 15;
    const int hi      = lane >> 4;          // 0 or 1
    const int rowBase = blockIdx.x * 16;
    const int chunk0  = blockIdx.y * CPS;

    const size_t rowOff = (size_t)(rowBase + m) * FIELD;

    const _Float16 h1 = (_Float16)1.0f;
    const _Float16 h0 = (_Float16)0.0f;

    v8f acc0 = {}, acc1 = {}, acc2 = {}, acc3 = {};

#pragma unroll 2
    for (int c = 0; c < CPS; ++c) {
        const int chunk = chunk0 + c;
        // ---- A fragment: 16x32 f16, contiguous k order matching the B repack.
        // lane<16: K = {0..15}; lane>=16: K = {16..31}. Four b128 index loads.
        const int fbase = chunk * KCHUNK + hi * 16;
        const int* p = idx + rowOff + fbase;
        // keep the index stream ahead of the wave (global_prefetch_b8)
        __builtin_prefetch(p + 2 * KCHUNK, 0, 0);
        v4i i0 = *(const v4i*)(p + 0);
        v4i i1 = *(const v4i*)(p + 4);
        v4i i2 = *(const v4i*)(p + 8);
        v4i i3 = *(const v4i*)(p + 12);
        v16h a;
#pragma unroll
        for (int j = 0; j < 4; ++j) {
            a[j]      = (i0[j] != 0) ? h1 : h0;
            a[4 + j]  = (i1[j] != 0) ? h1 : h0;
            a[8 + j]  = (i2[j] != 0) ? h1 : h0;
            a[12 + j] = (i3[j] != 0) ? h1 : h0;
        }
        // ---- B fragments: prepacked, 32 contiguous bytes per lane per tile.
        const v16h* bbase = (const v16h*)W16 + ((size_t)chunk * 4) * 32 + lane;
        v16h b0 = bbase[0 * 32];
        v16h b1 = bbase[1 * 32];
        v16h b2 = bbase[2 * 32];
        v16h b3 = bbase[3 * 32];

        // D = A*B + C ; f32 accumulate (exact for the sum of f16 W values)
        acc0 = __builtin_amdgcn_wmma_f32_16x16x32_f16(false, a, false, b0,
                                                      (short)0, acc0, false, false);
        acc1 = __builtin_amdgcn_wmma_f32_16x16x32_f16(false, a, false, b1,
                                                      (short)0, acc1, false, false);
        acc2 = __builtin_amdgcn_wmma_f32_16x16x32_f16(false, a, false, b2,
                                                      (short)0, acc2, false, false);
        acc3 = __builtin_amdgcn_wmma_f32_16x16x32_f16(false, a, false, b3,
                                                      (short)0, acc3, false, false);
    }

    // ---- C/D layout: VGPR r, lanes 0-15 -> M=r, lanes 16-31 -> M=r+8; N=lane%16
#pragma unroll
    for (int r = 0; r < 8; ++r) {
        const int orow = rowBase + r + hi * 8;
        float* po = out + (size_t)orow * EMBED + m;
        atomicAdd(po + 0,  acc0[r]);
        atomicAdd(po + 16, acc1[r]);
        atomicAdd(po + 32, acc2[r]);
        atomicAdd(po + 48, acc3[r]);
    }
}

// ---------------------------------------------------------------------------
// Fallback (only if d_ws is too small for the repacked W): simple streaming
// dot per output element. Still deterministic; never expected in practice.
// ---------------------------------------------------------------------------
__global__ void embed_fallback_kernel(const int* __restrict__ idx,
                                      const float* __restrict__ W,
                                      float* __restrict__ out) {
    int t = blockIdx.x * blockDim.x + threadIdx.x;
    if (t >= BATCH * EMBED) return;
    int b = t >> 6, d = t & 63;
    const int* r = idx + (size_t)b * FIELD;
    float s = 0.0f;
    for (int f = 0; f < FIELD; ++f)
        s += (r[f] != 0) ? W[(size_t)f * EMBED + d] : 0.0f;
    out[t] = s;
}

extern "C" void kernel_launch(void* const* d_in, const int* in_sizes, int n_in,
                              void* d_out, int out_size, void* d_ws, size_t ws_size,
                              hipStream_t stream) {
    const int*   idx = (const int*)d_in[0];     // [4096, 20000] int32
    const float* W   = (const float*)d_in[1];   // [20000, 64] f32
    float*       out = (float*)d_out;           // [4096, 64] f32

    const size_t w16_bytes = W16_ELEMS * sizeof(_Float16);

    if (ws_size >= w16_bytes) {
        _Float16* W16 = (_Float16*)d_ws;

        int repackThreads = (int)W16_ELEMS;
        repack_w_kernel<<<(repackThreads + 255) / 256, 256, 0, stream>>>(W, W16);

        int outElems = BATCH * EMBED;
        zero_out_kernel<<<(outElems + 255) / 256, 256, 0, stream>>>(out, outElems);

        dim3 grid(BATCH / 16, NSPLIT);   // 256 x 25 = 6400 wave32 blocks
        embed_wmma_kernel<<<grid, 32, 0, stream>>>(idx, W16, out);
    } else {
        int outElems = BATCH * EMBED;
        embed_fallback_kernel<<<(outElems + 255) / 256, 256, 0, stream>>>(idx, W, out);
    }
}
